// ExternalParisiNashRouter_85830626443514
// MI455X (gfx1250) — matrix operand order repaired and checked by reading, not compiled, not run on hardware
//
#include <hip/hip_runtime.h>
#include <hip/hip_bf16.h>
#include <math.h>

// ---------------------------------------------------------------------------
// MoE router forward for MI455X (gfx1250, wave32, WMMA).
// B=8 L=4096 D=1024 H=64 NB=16 K=2, TEMP=2.0, LN_EPS=1e-5
// HBM-bound: one 128MiB pass over x; GEMMs via v_wmma_f32_16x16x32_bf16.
// ---------------------------------------------------------------------------

typedef __attribute__((ext_vector_type(16))) __bf16         v16bf;
typedef __attribute__((ext_vector_type(8)))  float          v8f;
typedef __attribute__((ext_vector_type(8)))  unsigned short v8u16;

union ABfrag { v16bf bf; v8u16 u[2]; };

__device__ __forceinline__ unsigned short f2bf(float f) {
    // round-to-nearest-even f32 -> bf16
    unsigned int u = __float_as_uint(f);
    unsigned int r = u + 0x7FFFu + ((u >> 16) & 1u);
    return (unsigned short)(r >> 16);
}

// ---------------------------------------------------------------------------
// Kernel 0: pack W1 (1024x64) / W2 (64x16) into bf16 WMMA B-fragment order and
// zero the aux-loss accumulators. B-matrix 32x16 bf16 layout (ISA 7.12.2):
// lane = {khalf(1b), col(4b)}; lane holds 16 K values (kbase =
// kchunk*32 + 8*khalf; K kbase+0..7 then kbase+16..23), even/odd K packed low/high per dword. We store each
// lane's 16 bf16 contiguously -> one 32B load per fragment per lane.
// ---------------------------------------------------------------------------
__global__ __launch_bounds__(256) void router_prep(
    const float* __restrict__ W1, const float* __restrict__ W2,
    unsigned short* __restrict__ w1pk, unsigned short* __restrict__ w2pk,
    float* __restrict__ gcnt, float* __restrict__ gpsum)
{
    const int tid = blockIdx.x * 256 + threadIdx.x;   // 65536 threads total
    {
        const int j    = tid & 15;
        const int lane = (tid >> 4) & 31;
        const int kc   = (tid >> 9) & 31;
        const int nt   = (tid >> 14) & 3;
        const int Kidx = kc * 32 + ((j >> 3) << 4) + ((lane >> 4) << 3) + (j & 7);
        w1pk[tid] = f2bf(W1[Kidx * 64 + nt * 16 + (lane & 15)]);
    }
    if (tid < 1024) {
        const int j    = tid & 15;
        const int lane = (tid >> 4) & 31;
        const int kc   = (tid >> 9) & 1;
        const int Kidx = kc * 32 + ((j >> 3) << 4) + ((lane >> 4) << 3) + (j & 7);
        w2pk[tid] = f2bf(W2[Kidx * 16 + (lane & 15)]);
    }
    if (tid < 16) { gcnt[tid] = 0.0f; gpsum[tid] = 0.0f; }
}

// ---------------------------------------------------------------------------
// Kernel 1: main router. 16 tokens per workgroup of 256 threads (8 wave32s).
// ---------------------------------------------------------------------------
__global__ __launch_bounds__(256) void router_main(
    const float* __restrict__ x, const float* __restrict__ gamma,
    const float* __restrict__ beta,
    const unsigned short* __restrict__ w1pk,
    const unsigned short* __restrict__ w2pk,
    float* __restrict__ probs, float* __restrict__ idxo,
    float* __restrict__ wtso,
    float* __restrict__ gcnt, float* __restrict__ gpsum)
{
    // xn as bf16, row-major [m][k]: A-fragment rows are two b128 LDS loads.
    __shared__ unsigned short xnsh[16 * 1024];   // 32 KB
    __shared__ float          hred[4][8][32];    // 4 KB (K-half reduction)
    __shared__ unsigned short hsh[16 * 64];      // 2 KB (h as bf16)

    const int tid    = threadIdx.x;
    const int lane   = tid & 31;
    const int w      = tid >> 5;       // wave id 0..7
    const int laneHi = lane >> 4;      // half-warp select
    const int n16    = lane & 15;
    const int tokbase = blockIdx.x * 16;

    // ---- LayerNorm: each wave handles 2 tokens, x register-resident -------
    for (int tt = 0; tt < 2; ++tt) {
        const int tl = w * 2 + tt;
        const float4* xp = (const float4*)(x + (size_t)(tokbase + tl) * 1024);
        float4 v[8];
        float s = 0.0f, ss = 0.0f;
        #pragma unroll
        for (int j = 0; j < 8; ++j) {
            v[j] = xp[j * 32 + lane];
            s  += v[j].x + v[j].y + v[j].z + v[j].w;
            ss += v[j].x * v[j].x + v[j].y * v[j].y
                + v[j].z * v[j].z + v[j].w * v[j].w;
        }
        #pragma unroll
        for (int m = 16; m; m >>= 1) {
            s  += __shfl_xor(s, m);
            ss += __shfl_xor(ss, m);
        }
        const float mean = s * (1.0f / 1024.0f);
        const float var  = ss * (1.0f / 1024.0f) - mean * mean;
        const float rs   = rsqrtf(var + 1e-5f);
        #pragma unroll
        for (int j = 0; j < 8; ++j) {
            const float4 g = ((const float4*)gamma)[j * 32 + lane];
            const float4 b = ((const float4*)beta)[j * 32 + lane];
            ushort4 h;
            h.x = f2bf((v[j].x - mean) * rs * g.x + b.x);
            h.y = f2bf((v[j].y - mean) * rs * g.y + b.y);
            h.z = f2bf((v[j].z - mean) * rs * g.z + b.z);
            h.w = f2bf((v[j].w - mean) * rs * g.w + b.w);
            *(ushort4*)&xnsh[tl * 1024 + (j * 32 + lane) * 4] = h;
        }
    }
    __syncthreads();

    // ---- GEMM1: xn(16x1024) @ W1(1024x64). 4 N-tiles x 2 K-halves --------
    const int nt = w & 3;
    const int kh = w >> 2;
    v8f acc = {};
    const unsigned short* arow = &xnsh[n16 * 1024];
    for (int c = kh * 16; c < kh * 16 + 16; ++c) {
        ABfrag a, b;
        const v8u16* ap = (const v8u16*)(arow + c * 32);
        a.u[0] = ap[laneHi];         // K 0-7 / 8-15 of chunk
        a.u[1] = ap[2 + laneHi];     // K 16-23 / 24-31 of chunk
        const v8u16* bp =
            (const v8u16*)(w1pk + ((size_t)(nt * 32 + c) * 32 + lane) * 16);
        b.u[0] = bp[0];
        b.u[1] = bp[1];
        acc = __builtin_amdgcn_wmma_f32_16x16x32_bf16(
            false, a.bf, false, b.bf, (short)0, acc, false, false);
    }
    if (kh == 1) {
        #pragma unroll
        for (int r = 0; r < 8; ++r) hred[nt][r][lane] = acc[r];
    }
    __syncthreads();
    if (kh == 0) {
        // reduce K-halves, exact GELU, stage h as bf16 [m][hcol]
        #pragma unroll
        for (int r = 0; r < 8; ++r) {
            float hv = acc[r] + hred[nt][r][lane];
            hv = 0.5f * hv * (1.0f + erff(hv * 0.70710678118654752f));
            hsh[(r + 8 * laneHi) * 64 + nt * 16 + n16] = f2bf(hv);
        }
    }
    __syncthreads();

    // ---- GEMM2 + softmax + top-2 on wave 0 (wave-uniform branch) ----------
    if (w == 0) {
        v8f lac = {};
        #pragma unroll
        for (int c = 0; c < 2; ++c) {
            ABfrag a, b;
            const v8u16* ap = (const v8u16*)(&hsh[n16 * 64 + c * 32]);
            a.u[0] = ap[laneHi];
            a.u[1] = ap[2 + laneHi];
            const v8u16* bp = (const v8u16*)(w2pk + (size_t)(c * 32 + lane) * 16);
            b.u[0] = bp[0];
            b.u[1] = bp[1];
            lac = __builtin_amdgcn_wmma_f32_16x16x32_bf16(
                false, a.bf, false, b.bf, (short)0, lac, false, false);
        }
        float psum_acc = 0.0f, cnt_acc = 0.0f;
        #pragma unroll
        for (int r = 0; r < 8; ++r) {
            const float v = lac[r] * 0.5f;                 // /TEMP
            // softmax over the 16 lanes of this half-warp (one row each)
            float mx = v;
            #pragma unroll
            for (int m = 8; m; m >>= 1) mx = fmaxf(mx, __shfl_xor(mx, m));
            const float e = expf(v - mx);
            float sum = e;
            #pragma unroll
            for (int m = 8; m; m >>= 1) sum += __shfl_xor(sum, m);
            const float p = e / sum;
            const int tok = tokbase + r + 8 * laneHi;
            probs[(size_t)tok * 16 + n16] = p;
            // top-1 (ties -> lowest index, jax semantics)
            float v1 = p; int i1 = n16;
            #pragma unroll
            for (int m = 8; m; m >>= 1) {
                const float ov = __shfl_xor(v1, m);
                const int   oi = __shfl_xor(i1, m);
                if (ov > v1 || (ov == v1 && oi < i1)) { v1 = ov; i1 = oi; }
            }
            // top-2 (exclude i1)
            float v2 = (n16 == i1) ? -1.0f : p; int i2 = n16;
            #pragma unroll
            for (int m = 8; m; m >>= 1) {
                const float ov = __shfl_xor(v2, m);
                const int   oi = __shfl_xor(i2, m);
                if (ov > v2 || (ov == v2 && oi < i2)) { v2 = ov; i2 = oi; }
            }
            if (n16 == 0) {
                const float denom = v1 + v2 + 1e-8f;
                idxo[(size_t)tok * 2 + 0] = (float)i1;
                idxo[(size_t)tok * 2 + 1] = (float)i2;
                wtso[(size_t)tok * 2 + 0] = v1 / denom;
                wtso[(size_t)tok * 2 + 1] = v2 / denom;
            }
            psum_acc += p;
            cnt_acc  += (n16 == i1 ? 1.0f : 0.0f) + (n16 == i2 ? 1.0f : 0.0f);
        }
        atomicAdd(&gpsum[n16], psum_acc);
        atomicAdd(&gcnt[n16],  cnt_acc);
    }
}

// ---------------------------------------------------------------------------
// Kernel 2: aux_loss = NB * sum_b( counts_b/(B*L*K) * mean_probs_b )
// ---------------------------------------------------------------------------
__global__ void router_finalize(const float* __restrict__ gcnt,
                                const float* __restrict__ gpsum,
                                float* __restrict__ auxp)
{
    const int l = threadIdx.x;
    float t = 0.0f;
    if (l < 16)
        t = (gcnt[l] * (1.0f / 65536.0f)) * (gpsum[l] * (1.0f / 32768.0f));
    #pragma unroll
    for (int m = 16; m; m >>= 1) t += __shfl_xor(t, m);
    if (l == 0) auxp[0] = 16.0f * t;
}

// ---------------------------------------------------------------------------
extern "C" void kernel_launch(void* const* d_in, const int* in_sizes, int n_in,
                              void* d_out, int out_size, void* d_ws, size_t ws_size,
                              hipStream_t stream) {
    const float* x  = (const float*)d_in[0];
    const float* g  = (const float*)d_in[1];
    const float* bb = (const float*)d_in[2];
    const float* W1 = (const float*)d_in[3];
    const float* W2 = (const float*)d_in[4];

    float* out   = (float*)d_out;
    float* probs = out;                 // 8*4096*16 = 524288
    float* idxo  = out + 524288;        // 8*4096*2  =  65536 (as float)
    float* auxp  = out + 589824;        // 1
    float* wtso  = out + 589825;        // 65536

    unsigned short* w1pk = (unsigned short*)d_ws;          // 131072 B
    unsigned short* w2pk = w1pk + 65536;                   //   2048 B
    float* gcnt  = (float*)((char*)d_ws + 133120);         //     64 B
    float* gpsum = gcnt + 16;                              //     64 B

    router_prep    <<<256,  256, 0, stream>>>(W1, W2, w1pk, w2pk, gcnt, gpsum);
    router_main    <<<2048, 256, 0, stream>>>(x, g, bb, w1pk, w2pk,
                                              probs, idxo, wtso, gcnt, gpsum);
    router_finalize<<<1,    32,  0, stream>>>(gcnt, gpsum, auxp);
}
